// WeightedSigLipLoss_44315472560540
// MI455X (gfx1250) — matrix operand order repaired and checked by reading, not compiled, not run on hardware
//
#include <hip/hip_runtime.h>
#include <hip/hip_bf16.h>
#include <math.h>

#define N_ROWS 8192
#define DIM    512
#define NCLS   6
#define TPAD   32
#define WAVES_PER_BLOCK 8
#define COL_SPLITS 8
#define COLS_PER_SPLIT (N_ROWS / COL_SPLITS)   // 1024 columns per split
#define ROWP   520                              // padded LDS row stride in f16 (1040B = 65*16B)

typedef _Float16 v16h __attribute__((ext_vector_type(16)));
typedef _Float16 v8h  __attribute__((ext_vector_type(8)));
typedef float    v8f  __attribute__((ext_vector_type(8)));

// ---- workspace layout (d_ws) ----
// [0,32768)          : float acc[8192]
// [32768,32772)      : float sumw
// [65536, +8MB)      : _Float16 fh[8192][512]  normalized f16 features
// [65536+8MB,+512KB) : _Float16 tpad[8192][32] zero-padded f16 targets
#define WS_FH_OFF   (65536)
#define WS_TP_OFF   (WS_FH_OFF + (size_t)N_ROWS * DIM * 2)

#if __has_builtin(__builtin_amdgcn_s_wait_asynccnt)
#define WAIT_ASYNC(n) __builtin_amdgcn_s_wait_asynccnt(n)
#else
#define WAIT_ASYNC(n) asm volatile("s_wait_asynccnt %0" :: "i"(n))
#endif

// Low 32 bits of a generic (flat) LDS pointer are the LDS byte offset
static __device__ __forceinline__ uint32_t lds_off(const void* p) {
  return (uint32_t)(uintptr_t)p;
}

// CDNA5 async copy global->LDS, 16 bytes per lane, ASYNCcnt-tracked.
static __device__ __forceinline__ void async_copy_b128(uint32_t lds_byte_addr,
                                                       const void* gptr) {
  asm volatile("global_load_async_to_lds_b128 %0, %1, off"
               :: "v"(lds_byte_addr), "v"((uint64_t)(uintptr_t)gptr)
               : "memory");
}

__global__ void init_kernel(float* __restrict__ accw) {
  int i = blockIdx.x * blockDim.x + threadIdx.x;
  if (i < N_ROWS + 1) accw[i] = 0.0f;   // acc[8192] then sumw, contiguous
}

// One block (128 threads) per row: L2-normalize -> f16 row; pad targets; sum weights.
__global__ __launch_bounds__(128) void prep_kernel(
    const float* __restrict__ features, const float* __restrict__ targets,
    const float* __restrict__ weights,
    _Float16* __restrict__ fh, _Float16* __restrict__ tpad,
    float* __restrict__ sumw) {
  const int row = blockIdx.x;
  const int tid = threadIdx.x;
  __shared__ float red[128];
  const float* frow = features + row * DIM;
  float s = 0.0f;
  for (int k = tid; k < DIM; k += 128) { float v = frow[k]; s = fmaf(v, v, s); }
  red[tid] = s;
  __syncthreads();
  for (int off = 64; off > 0; off >>= 1) {
    if (tid < off) red[tid] += red[tid + off];
    __syncthreads();
  }
  const float inv = 1.0f / fmaxf(sqrtf(red[0]), 1e-12f);
  _Float16* orow = fh + row * DIM;
  for (int k = tid; k < DIM; k += 128) orow[k] = (_Float16)(frow[k] * inv);
  if (tid < TPAD) {
    float tv = (tid < NCLS) ? targets[row * NCLS + tid] : 0.0f;
    tpad[row * TPAD + tid] = (_Float16)tv;
  }
  if (tid == 0) atomicAdd(sumw, weights[row]);
}

static __device__ __forceinline__ v16h cat16(v8h lo, v8h hi) {
  return __builtin_shufflevector(lo, hi, 0,1,2,3,4,5,6,7,8,9,10,11,12,13,14,15);
}

// Main fused kernel. Block = 8 waves = 128 rows; the 16-column B panel (16KB)
// is staged once per block into LDS via async copies and shared by all 8 waves.
// Inner k-loop uses register double-buffering of the B fragment so ds_load of
// step k+1 overlaps the WMMA of step k (no ds_load -> wait -> wmma serialization).
__global__ __launch_bounds__(256) void siglip_main(
    const _Float16* __restrict__ fh, const _Float16* __restrict__ tpad,
    const float* __restrict__ tparam, const float* __restrict__ bparam,
    float* __restrict__ acc) {
  // Double-buffered B panel: 16 rows x 512 f16, padded row stride 520 f16.
  __shared__ _Float16 ldsb[2][16 * ROWP];

  const int tid  = threadIdx.x;
  const int lane = tid & 31;
  const int wave = tid >> 5;
  const int half = lane >> 4;
  const int mr   = lane & 15;
  const int m0   = (blockIdx.x * WAVES_PER_BLOCK + wave) * 16;
  const int n_begin = blockIdx.y * COLS_PER_SPLIT;
  const int n_end   = n_begin + COLS_PER_SPLIT;

  const float tval = tparam[0];
  const float bval = bparam[0];

  // Per-thread async-copy slice: row r = tid>>4, chunks c = (tid&15)+16*i, i=0..3.
  const int cp_r = tid >> 4;
  const int cp_c = tid & 15;

  // Preload target A tile for this wave's 16 rows (K padded to 32).
  v16h ta;
  {
    const _Float16* base = tpad + (m0 + mr) * TPAD;
    v8h lo = *(const v8h*)(base + half * 8);
    v8h hi = *(const v8h*)(base + 16 + half * 8);
    ta = cat16(lo, hi);
  }

  // Preload all 16 A k-tiles (16x32 f16 each) for this wave's rows: 128 VGPRs.
  const _Float16* arow = fh + (m0 + mr) * DIM;
  v16h atile[16];
#pragma unroll
  for (int kk = 0; kk < 16; ++kk) {
    v8h alo = *(const v8h*)(arow + kk * 32 + half * 8);
    v8h ahi = *(const v8h*)(arow + kk * 32 + 16 + half * 8);
    atile[kk] = cat16(alo, ahi);
  }

  float rowsum[8];
#pragma unroll
  for (int r = 0; r < 8; ++r) rowsum[r] = 0.0f;

  // Prologue: prefetch first B panel into buffer 0 (4 async B128 per thread).
  {
    const _Float16* gbase = fh + (size_t)(n_begin + cp_r) * DIM + cp_c * 8;
    _Float16* lbase = &ldsb[0][cp_r * ROWP + cp_c * 8];
#pragma unroll
    for (int i = 0; i < 4; ++i)
      async_copy_b128(lds_off(lbase + i * 128), gbase + i * 128);
  }

  int buf = 0;
  for (int n0 = n_begin; n0 < n_end; n0 += 16) {
    // Prefetch next panel into the other buffer, then drain current panel.
    if (n0 + 16 < n_end) {
      const _Float16* gbase = fh + (size_t)(n0 + 16 + cp_r) * DIM + cp_c * 8;
      _Float16* lbase = &ldsb[buf ^ 1][cp_r * ROWP + cp_c * 8];
#pragma unroll
      for (int i = 0; i < 4; ++i)
        async_copy_b128(lds_off(lbase + i * 128), gbase + i * 128);
      WAIT_ASYNC(4);   // in-order completion: current panel's 4 ops are done
    } else {
      WAIT_ASYNC(0);
    }
    __syncthreads();   // every wave's slice of the current panel has landed

    // Issue the (independent) agree-tile global load early so it completes
    // behind the 16 sim WMMAs.
    v8h tblo, tbhi;
    {
      const _Float16* tb = tpad + (n0 + mr) * TPAD + half * 8;
      tblo = *(const v8h*)(tb);
      tbhi = *(const v8h*)(tb + 16);
    }

    // ---- sim tile: 16 x v_wmma_f32_16x16x32_f16, B from LDS, SW-pipelined ----
    // B layout: lane L holds column n0+mr, K at k0 + half*8 + [0..7] and k0 + 16 + half*8 + [0..7].
    const _Float16* bbase = &ldsb[buf][mr * ROWP + half * 8];
    v16h bcur;
    {
      v8h blo = *(const v8h*)(bbase);
      v8h bhi = *(const v8h*)(bbase + 16);
      bcur = cat16(blo, bhi);
    }
    v8f sim = {};
#pragma unroll
    for (int kk = 0; kk < 16; ++kk) {
      v16h bnext = bcur;
      if (kk < 15) {   // load fragment k+1 BEFORE consuming fragment k
        v8h nlo = *(const v8h*)(bbase + (kk + 1) * 32);
        v8h nhi = *(const v8h*)(bbase + (kk + 1) * 32 + 16);
        bnext = cat16(nlo, nhi);
      }
      sim = __builtin_amdgcn_wmma_f32_16x16x32_f16(
          false, atile[kk], false, bcur, (short)0, sim, false, false);
      bcur = bnext;
    }

    // ---- agree tile: one WMMA on zero-padded targets ----
    v8f g = {};
    g = __builtin_amdgcn_wmma_f32_16x16x32_f16(
        false, ta, false, cat16(tblo, tbhi), (short)0, g, false, false);

    // ---- fused epilogue: VGPR r, lane L -> M = m0 + half*8 + r, N = n0 + mr ----
#pragma unroll
    for (int r = 0; r < 8; ++r) {
      float agree = fmaf(2.0f, g[r], -1.0f);
      float e  = agree * fmaf(tval, sim[r], bval);
      float sp = fmaxf(e, 0.0f) + __logf(1.0f + __expf(-fabsf(e)));
      rowsum[r] += sp;
    }
    __syncthreads();   // all waves done reading this buffer before it is refilled
    buf ^= 1;
  }

  // Reduce across the 16 lanes sharing the same M (lanes differ only in N).
#pragma unroll
  for (int r = 0; r < 8; ++r) {
    float v = rowsum[r];
    v += __shfl_xor(v, 1);
    v += __shfl_xor(v, 2);
    v += __shfl_xor(v, 4);
    v += __shfl_xor(v, 8);
    rowsum[r] = v;
  }
  if (mr == 0) {
#pragma unroll
    for (int r = 0; r < 8; ++r)
      atomicAdd(&acc[m0 + half * 8 + r], rowsum[r]);
  }
}

__global__ void finalize_kernel(const float* __restrict__ acc,
                                const float* __restrict__ weights,
                                const float* __restrict__ sumw,
                                float* __restrict__ out) {
  int i = blockIdx.x * blockDim.x + threadIdx.x;
  // loss = (rowsum/N) * w / (sumw/N) = rowsum * w / sumw
  if (i < N_ROWS) out[i] = acc[i] * weights[i] / sumw[0];
}

extern "C" void kernel_launch(void* const* d_in, const int* in_sizes, int n_in,
                              void* d_out, int out_size, void* d_ws, size_t ws_size,
                              hipStream_t stream) {
  (void)in_sizes; (void)n_in; (void)out_size; (void)ws_size;
  // setup_inputs order: logits(unused), targets, features, weights, t, b
  const float* targets  = (const float*)d_in[1];
  const float* features = (const float*)d_in[2];
  const float* weights  = (const float*)d_in[3];
  const float* tparam   = (const float*)d_in[4];
  const float* bparam   = (const float*)d_in[5];
  float* out = (float*)d_out;

  char* ws = (char*)d_ws;
  float*     acc  = (float*)ws;            // [8192] + sumw right after
  float*     sumw = acc + N_ROWS;
  _Float16*  fh   = (_Float16*)(ws + WS_FH_OFF);
  _Float16*  tp   = (_Float16*)(ws + WS_TP_OFF);

  init_kernel<<<(N_ROWS + 1 + 255) / 256, 256, 0, stream>>>(acc);
  prep_kernel<<<N_ROWS, 128, 0, stream>>>(features, targets, weights, fh, tp, sumw);

  dim3 grid(N_ROWS / (16 * WAVES_PER_BLOCK), COL_SPLITS);   // 64 x 8 = 512 blocks
  siglip_main<<<grid, 256, 0, stream>>>(fh, tp, tparam, bparam, acc);

  finalize_kernel<<<(N_ROWS + 255) / 256, 256, 0, stream>>>(acc, weights, sumw, out);
}
